// CoattentionNet_28458453303692
// MI455X (gfx1250) — compile-verified, matrix operand order checked by reading, not run on hardware
//
#include <hip/hip_runtime.h>
#include <hip/hip_bf16.h>

typedef __bf16 bf16_t;
typedef __attribute__((ext_vector_type(8)))  __bf16 v8bf;
typedef __attribute__((ext_vector_type(16))) __bf16 v16bf;
typedef __attribute__((ext_vector_type(8)))  float  v8f;
typedef __attribute__((ext_vector_type(4)))  unsigned v4u;
typedef __attribute__((ext_vector_type(8)))  int      v8i_;
typedef __attribute__((ext_vector_type(4)))  int      v4i_;

#define BB 512
#define LL 256
#define DD 256
#define OO 1000
#define EP 264   // padded LDS row stride (bf16 elems): 528B = 132 dwords -> bank-staggered rows

// native gfx1250 bf16 converts (single v_cvt op instead of bit-trick VALU chains)
static __device__ __forceinline__ bf16_t f2bf(float f) { return (bf16_t)f; }
static __device__ __forceinline__ float  bf2f(bf16_t b) { return (float)b; }

// One 16-wide bf16 fragment for A/B operands of v_wmma_f32_16x16x32_bf16.
// Per ISA layout, lane needs row[k0+kb .. +7] and row[k0+16+kb .. +7], kb = (lane>>4)*8.
// Both runs are 16B-aligned contiguous loads (b128).
static __device__ __forceinline__ v16bf frag_row(const bf16_t* __restrict__ row, int kb) {
  v8bf lo = *reinterpret_cast<const v8bf*>(row + kb);
  v8bf hi = *reinterpret_cast<const v8bf*>(row + 16 + kb);
  return __builtin_shufflevector(lo, hi, 0,1,2,3,4,5,6,7,8,9,10,11,12,13,14,15);
}

// ---------------- gather embedding rows, convert f32 -> bf16 ----------------
__global__ void __launch_bounds__(256) gather_emb(const int* __restrict__ idx,
                                                  const float* __restrict__ emb,
                                                  bf16_t* __restrict__ Ebf) {
  int row  = blockIdx.x * 8 + (threadIdx.x >> 5);
  int lane = threadIdx.x & 31;
  long long e = (long long)idx[row] * DD;
  const float4* src = reinterpret_cast<const float4*>(emb + e) + lane * 2;
  float4 a = src[0], b = src[1];
  v8bf o;
  o[0]=f2bf(a.x); o[1]=f2bf(a.y); o[2]=f2bf(a.z); o[3]=f2bf(a.w);
  o[4]=f2bf(b.x); o[5]=f2bf(b.y); o[6]=f2bf(b.z); o[7]=f2bf(b.w);
  *reinterpret_cast<v8bf*>(Ebf + (size_t)row * DD + lane * 8) = o;
}

// W_b [k][n] f32 -> W_bT [n][k] bf16  (so B-fragments are contiguous row reads)
__global__ void __launch_bounds__(256) conv_wbt(const float* __restrict__ wb,
                                                bf16_t* __restrict__ wbt) {
  int t = blockIdx.x * 256 + threadIdx.x;   // t = n*256 + k
  int n = t >> 8, k = t & 255;
  wbt[t] = f2bf(wb[k * DD + n]);
}

// lin_w [O][D] f32 -> bf16 (already the right orientation for B operand)
__global__ void __launch_bounds__(256) conv_linw(const float* __restrict__ w,
                                                 bf16_t* __restrict__ wbf, int n) {
  int t = blockIdx.x * 256 + threadIdx.x;
  if (t < n) wbf[t] = f2bf(w[t]);
}

// ---------------- fused per-batch co-attention ----------------
// One 256-thread block (8 waves) per batch. LDS: E (132KB) + T (132KB) + softmax scratch,
// ~272KB of the 320KB WGP LDS. E tile is DMA'd in by the Tensor Data Mover (padded rows),
// tracked with TENSORcnt. Phase1: T = E*Wb (WMMA). Phase2: S = T*E^T (WMMA) with register
// row-max (tanh deferred: monotonic). Phase3: softmax + f_w = a^T E.
__global__ void __launch_bounds__(256, 1)
coattn_fused(const bf16_t* __restrict__ Ebf, const bf16_t* __restrict__ wbt,
             bf16_t* __restrict__ fw) {
  extern __shared__ char smem[];
  bf16_t* Es   = reinterpret_cast<bf16_t*>(smem);        // 256 x EP
  bf16_t* Ts   = Es + 256 * EP;                          // 256 x EP
  float*  sval = reinterpret_cast<float*>(Ts + 256 * EP);// 256
  float*  red  = sval + 256;                             // 256

  const int tid = threadIdx.x, lane = tid & 31, wv = tid >> 5;
  const int nlane = lane & 15, halfsel = lane >> 4, kb = halfsel * 8;
  const bf16_t* Eg = Ebf + (size_t)blockIdx.x * LL * DD;

  // warm L2/L0 for the shared W_bT while the E tile is in flight (global_prefetch_b8)
#pragma unroll
  for (int i = 0; i < 4; ++i)
    __builtin_prefetch((const char*)wbt + ((size_t)tid * 4 + i) * 128, 0, 1);

#if __has_builtin(__builtin_amdgcn_tensor_load_to_lds)
  // TDM: DMA the 256x256 bf16 tile global->LDS with 16B padding appended per 512B row
  // (pad_interval code 6 = 128 dwords, pad_amount code 3 = 4 dwords) -> EP=264 layout.
  if (wv == 0) {
    unsigned long long ga = (unsigned long long)(const void*)Eg;
    v4u g0;
    g0[0] = 1u;                                         // count=1 valid user D#
    g0[1] = 0u;                                         // lds_addr = base of dynamic LDS
    g0[2] = (unsigned)(ga & 0xffffffffu);               // global_addr[31:0]
    g0[3] = (unsigned)((ga >> 32) & 0x01ffffffu) | (2u << 30);  // addr[56:32] | type=2
    v8i_ g1;
    g1[0] = (int)((1u << 16) | (1u << 20) | (6u << 22) | (3u << 25)); // 2B, pad en/int/amt
    g1[1] = (int)(256u << 16);   // tensor_dim0 = 256
    g1[2] = (int)(256u << 16);   // tensor_dim1 = 256 (dim0 hi bits = 0)
    g1[3] = (int)(256u << 16);   // tile_dim0 = 256
    g1[4] = 256;                 // tile_dim1 = 256, tile_dim2 = 0
    g1[5] = 256;                 // tensor_dim0_stride = 256
    g1[6] = 0;
    g1[7] = 0;                   // tensor_dim1_stride unused (2D)
    v4i_ gz = {};                // groups 2/3 unused for 2D tiles
    v8i_ gz8 = {};               // extra group (6-arg toolchain variant)
    __builtin_amdgcn_tensor_load_to_lds(g0, g1, gz, gz, gz8, 0);
    __builtin_amdgcn_s_wait_tensorcnt(0);
  }
#else
  // fallback: cooperative copy, one row per wave iteration, 32 lanes x 16B
  for (int r = wv * 32; r < wv * 32 + 32; ++r) {
    v8bf v = *reinterpret_cast<const v8bf*>(Eg + r * DD + lane * 8);
    *reinterpret_cast<v8bf*>(Es + r * EP + lane * 8) = v;
  }
#endif
  __syncthreads();

  // Phase 1: T = E @ W_b ; wave w owns row-tiles {2w, 2w+1} x all 16 col-tiles
  for (int tmi = 0; tmi < 2; ++tmi) {
    const int tm = wv * 2 + tmi;
    const bf16_t* arow = Es + (tm * 16 + nlane) * EP;
    v16bf afr[8];
#pragma unroll
    for (int ks = 0; ks < 8; ++ks) afr[ks] = frag_row(arow + ks * 32, kb);
    for (int tn = 0; tn < 16; ++tn) {
      const bf16_t* brow = wbt + (size_t)(tn * 16 + nlane) * DD; // B[k][n] = Wb[k][n]
      v8f acc = {};
#pragma unroll
      for (int ks = 0; ks < 8; ++ks) {
        v16bf bfr = frag_row(brow + ks * 32, kb);
        acc = __builtin_amdgcn_wmma_f32_16x16x32_bf16(false, afr[ks], false, bfr,
                                                      (short)0, acc, false, false);
      }
#pragma unroll
      for (int r = 0; r < 8; ++r)
        Ts[(tm * 16 + r + 8 * halfsel) * EP + tn * 16 + nlane] = f2bf(acc[r]);
    }
  }
  __syncthreads();

  // Phase 2: S = T @ E^T with running row-max (tanh applied later; it's monotonic)
  for (int tmi = 0; tmi < 2; ++tmi) {
    const int tm = wv * 2 + tmi;
    const bf16_t* arow = Ts + (tm * 16 + nlane) * EP;
    v16bf afr[8];
#pragma unroll
    for (int ks = 0; ks < 8; ++ks) afr[ks] = frag_row(arow + ks * 32, kb);
    float rm[8];
#pragma unroll
    for (int r = 0; r < 8; ++r) rm[r] = -3.0e38f;
    for (int tn = 0; tn < 16; ++tn) {
      const bf16_t* brow = Es + (tn * 16 + nlane) * EP;  // B[k][n] = E[n][k]
      v8f acc = {};
#pragma unroll
      for (int ks = 0; ks < 8; ++ks) {
        v16bf bfr = frag_row(brow + ks * 32, kb);
        acc = __builtin_amdgcn_wmma_f32_16x16x32_bf16(false, afr[ks], false, bfr,
                                                      (short)0, acc, false, false);
      }
#pragma unroll
      for (int r = 0; r < 8; ++r) rm[r] = fmaxf(rm[r], acc[r]);
    }
    // reduce across the 16-lane N-group (stays within each half of the wave32)
#pragma unroll
    for (int r = 0; r < 8; ++r) {
      float v = rm[r];
      for (int off = 1; off < 16; off <<= 1) v = fmaxf(v, __shfl_xor(v, off, 32));
      rm[r] = v;
    }
    if (nlane == 0) {
#pragma unroll
      for (int r = 0; r < 8; ++r) sval[tm * 16 + r + 8 * halfsel] = rm[r];
    }
  }
  __syncthreads();

  // Phase 3: softmax over L of tanh(rowmax), then f_w[d] = sum_l a[l]*E[l][d]
  float tv = tanhf(sval[tid]);
  red[tid] = tv;
  __syncthreads();
  for (int s = 128; s > 0; s >>= 1) {
    if (tid < s) red[tid] = fmaxf(red[tid], red[tid + s]);
    __syncthreads();
  }
  float mx = red[0];
  __syncthreads();
  float ex = __expf(tv - mx);
  red[tid] = ex;
  __syncthreads();
  for (int s = 128; s > 0; s >>= 1) {
    if (tid < s) red[tid] += red[tid + s];
    __syncthreads();
  }
  float a = ex / red[0];
  __syncthreads();
  sval[tid] = a;
  __syncthreads();
  float acc = 0.f;
  for (int l = 0; l < 256; ++l) acc += sval[l] * bf2f(Es[l * EP + tid]);
  fw[(size_t)blockIdx.x * DD + tid] = f2bf(acc);
}

// ---------------- out = f_w @ lin_w^T + lin_b   (512 x 1000, K=256) ----------------
// 32 x 63 tiles of 16x16, one tile per wave. Clamp OOB B-rows (their D-columns are
// never stored: D-column n depends only on B-column n).
__global__ void __launch_bounds__(256)
final_linear(const bf16_t* __restrict__ fw, const bf16_t* __restrict__ lw,
             const float* __restrict__ lb, float* __restrict__ out) {
  const int lane = threadIdx.x & 31, wv = threadIdx.x >> 5;
  const int tile = blockIdx.x * 8 + wv;                  // 2016 tiles total
  const int tm = tile / 63, tn = tile % 63;
  const int nlane = lane & 15, halfsel = lane >> 4, kb = halfsel * 8;
  const int n = tn * 16 + nlane;
  const bf16_t* arow = fw + (size_t)(tm * 16 + nlane) * DD;
  const int nc = n < OO ? n : OO - 1;
  const bf16_t* brow = lw + (size_t)nc * DD;
  v8f acc = {};
#pragma unroll
  for (int ks = 0; ks < 8; ++ks) {
    v16bf afr = frag_row(arow + ks * 32, kb);
    v16bf bfr = frag_row(brow + ks * 32, kb);
    acc = __builtin_amdgcn_wmma_f32_16x16x32_bf16(false, afr, false, bfr,
                                                  (short)0, acc, false, false);
  }
  if (n < OO) {
    float bias = lb[n];
#pragma unroll
    for (int r = 0; r < 8; ++r) {
      int m = tm * 16 + r + 8 * halfsel;
      out[(size_t)m * OO + n] = acc[r] + bias;
    }
  }
}

extern "C" void kernel_launch(void* const* d_in, const int* in_sizes, int n_in,
                              void* d_out, int out_size, void* d_ws, size_t ws_size,
                              hipStream_t stream) {
  const int*   idx = (const int*)d_in[0];     // input_sentence [B,L]
  const float* emb = (const float*)d_in[1];   // emb_weight [V,D]
  const float* wb  = (const float*)d_in[2];   // W_b [D,D]
  const float* lw  = (const float*)d_in[3];   // lin_w [O,D]
  const float* lb  = (const float*)d_in[4];   // lin_b [O]
  float* out = (float*)d_out;                 // [B,O] f32

  char* ws = (char*)d_ws;
  bf16_t* Ebf  = (bf16_t*)ws; ws += (size_t)BB * LL * DD * 2;   // 64 MiB
  bf16_t* wbt  = (bf16_t*)ws; ws += (size_t)DD * DD * 2;        // 128 KiB
  bf16_t* lwbf = (bf16_t*)ws; ws += (size_t)OO * DD * 2;        // 500 KiB
  bf16_t* fwv  = (bf16_t*)ws; ws += (size_t)BB * DD * 2;        // 256 KiB

  gather_emb<<<BB * LL / 8, 256, 0, stream>>>(idx, emb, Ebf);
  conv_wbt<<<DD * DD / 256, 256, 0, stream>>>(wb, wbt);
  conv_linw<<<(OO * DD + 255) / 256, 256, 0, stream>>>(lw, lwbf, OO * DD);

  const size_t smem = (size_t)2 * 256 * EP * sizeof(bf16_t) + 2 * 256 * sizeof(float); // 272384 B
  coattn_fused<<<BB, 256, smem, stream>>>(Ebf, wbt, fwv);

  final_linear<<<(32 * 63) / 8, 256, 0, stream>>>(fwv, lwbf, lb, out);
}
